// EnhancedEncoderLayer_87694642249911
// MI455X (gfx1250) — matrix operand order repaired and checked by reading, not compiled, not run on hardware
//
#include <hip/hip_runtime.h>
#include <hip/hip_bf16.h>
#include <math.h>

#define BB   4
#define SS   1024
#define DD   1024
#define HH   16
#define RR   64
#define DFFX 4096
#define DH   64
#define KK_TOP 204   // int(1024*0.2)

#define USE_ASYNC_TLDS 1

typedef __attribute__((ext_vector_type(16))) __bf16 v16bf;
typedef __attribute__((ext_vector_type(8)))  __bf16 v8bf;
typedef __attribute__((ext_vector_type(8)))  float  v8f;

#define WMMA_BF16(a, b, c) \
  __builtin_amdgcn_wmma_f32_16x16x32_bf16(false, (a), false, (b), (short)0, (c), false, false)

// ---------------------------------------------------------------------------
// Fragment loaders (ISA 7.12.2 layouts, wave32).
// A (16x32, MxK): lane<16 -> M=lane, K={k0..k0+7, k0+16..k0+23};
//                 lane>=16 -> M=lane-16, K={k0+8..15, k0+24..31}.
// B (32x16, KxN): lane -> N=lane&15, K same as A (k0+(lane>>4)*8 +0..7, +16..23).
// C/D (16x16 f32): lane -> N=lane&15, VGPR r -> M = r + (lane>>4)*8.
// All loads below are contiguous 16B/32B vectors.
// ---------------------------------------------------------------------------
static __device__ __forceinline__ v16bf load_a_frag(const __bf16* A, int lda, int row,
                                                    int k0, int half) {
  const __bf16* p = A + (size_t)row * lda + k0 + half * 8;
  v8bf lo = *(const v8bf*)p;
  v8bf hi = *(const v8bf*)(p + 16);
  return __builtin_shufflevector(lo, hi, 0, 1, 2, 3, 4, 5, 6, 7,
                                 8, 9, 10, 11, 12, 13, 14, 15);
}

// W stored [N,K] row-major: contiguous along K.
static __device__ __forceinline__ v16bf load_b_nt(const __bf16* W, int ldw, int n,
                                                  int k0, int half) {
  const __bf16* p = W + (size_t)n * ldw + k0 + half * 8;
  v8bf lo = *(const v8bf*)p;
  v8bf hi = *(const v8bf*)(p + 16);
  return __builtin_shufflevector(lo, hi, 0, 1, 2, 3, 4, 5, 6, 7,
                                 8, 9, 10, 11, 12, 13, 14, 15);
}

// ---------------------------------------------------------------------------
// f32 -> bf16 convert
// ---------------------------------------------------------------------------
__global__ void cvt_f32_bf16(const float* __restrict__ in, __bf16* __restrict__ out, int n) {
  int i = blockIdx.x * blockDim.x + threadIdx.x;
  int stride = gridDim.x * blockDim.x;
  for (; i < n; i += stride) out[i] = (__bf16)in[i];
}

// ---------------------------------------------------------------------------
// Generic bf16 WMMA GEMM:  C[M,N] = scale * (A[M,K] x W[N,K]^T) + bias, opt ReLU.
// 8 waves / block; each wave computes a 32x64 tile: 2 A-frags x 4 B-frags,
// 8 WMMAs per 32-K step (each B-fragment reused twice).
// ---------------------------------------------------------------------------
__global__ void gemm_bf16(const __bf16* __restrict__ A, const __bf16* __restrict__ W,
                          const float* __restrict__ bias, float scale,
                          float* __restrict__ Cf, __bf16* __restrict__ Cb,
                          int M, int N, int K, int lda, int ldw, int ldc, int relu) {
  const int lane = threadIdx.x & 31;
  const int w    = threadIdx.x >> 5;
  const int l16  = lane & 15;
  const int half = lane >> 4;
  const int tiles_n = N >> 6;
  const int wid = blockIdx.x * 8 + w;
  const int tile_m = wid / tiles_n;
  const int tile_n = wid - tile_m * tiles_n;
  if (tile_m * 32 >= M) return;  // wave-uniform exit: EXEC stays all-ones for WMMA
  const int m0 = tile_m * 32, n0 = tile_n * 64;

  v8f acc[2][4] = {};
  for (int k0 = 0; k0 < K; k0 += 32) {
    v16bf a0 = load_a_frag(A, lda, m0 + l16, k0, half);
    v16bf a1 = load_a_frag(A, lda, m0 + 16 + l16, k0, half);
#pragma unroll
    for (int t = 0; t < 4; ++t) {
      v16bf b = load_b_nt(W, ldw, n0 + t * 16 + l16, k0, half);
      acc[0][t] = WMMA_BF16(a0, b, acc[0][t]);
      acc[1][t] = WMMA_BF16(a1, b, acc[1][t]);
    }
    __builtin_prefetch((const void*)(A + (size_t)(m0 + l16) * lda + k0 + 64), 0, 1);
  }

#pragma unroll
  for (int s = 0; s < 2; ++s)
#pragma unroll
    for (int t = 0; t < 4; ++t) {
      const int n = n0 + t * 16 + l16;
      const float bv = bias ? bias[n] : 0.0f;
#pragma unroll
      for (int r = 0; r < 8; ++r) {
        const int m = m0 + s * 16 + half * 8 + r;
        float v = acc[s][t][r] * scale + bv;
        if (relu) v = fmaxf(v, 0.0f);
        if (Cf) Cf[(size_t)m * ldc + n] = v;
        if (Cb) Cb[(size_t)m * ldc + n] = (__bf16)v;
      }
    }
}

// ---------------------------------------------------------------------------
// Transpose the per-head V slice of qkv into vT[b,h][dh][S] (bf16) so the
// attention PV B-fragments are contiguous. 64x64 tile staged through LDS.
// Global->LDS copy uses GLOBAL_LOAD_ASYNC_TO_LDS_B128 (ASYNCcnt-tracked,
// no VGPR round trip), then s_wait_asynccnt + barrier before readback.
// ---------------------------------------------------------------------------
__global__ void transpose_v_heads(const __bf16* __restrict__ qkv, __bf16* __restrict__ vT) {
  __shared__ __align__(16) __bf16 tile[64][72];   // 144B row pitch (16B multiple)
  const int tid = threadIdx.x;
  const int bh = blockIdx.x;
  const int s0 = blockIdx.y * 64;
  const int b = bh / HH, h = bh % HH;
  const __bf16* vsrc = qkv + ((size_t)b * SS + s0) * (3 * DD) + 2 * DD + h * DH;

#if USE_ASYNC_TLDS
#pragma unroll
  for (int i = 0; i < 2; ++i) {
    const int flat = tid + i * 256;       // 512 x 16B = 64 rows x 128B
    const int srow = flat >> 3;
    const int c8   = (flat & 7) * 8;
    const __bf16* gp = vsrc + (size_t)srow * (3 * DD) + c8;
    const unsigned ldsoff = (unsigned)(uintptr_t)(&tile[srow][c8]);
    asm volatile("global_load_async_to_lds_b128 %0, %1, off"
                 :: "v"(ldsoff), "v"(gp) : "memory");
  }
  asm volatile("s_wait_asynccnt 0x0" ::: "memory");
#else
#pragma unroll
  for (int i = 0; i < 2; ++i) {
    const int flat = tid + i * 256;
    const int srow = flat >> 3;
    const int c8   = (flat & 7) * 8;
    *(v8bf*)(&tile[srow][c8]) = *(const v8bf*)(vsrc + (size_t)srow * (3 * DD) + c8);
  }
#endif
  __syncthreads();

  __bf16* dst = vT + (size_t)bh * DH * SS + s0;   // row d, stride SS
#pragma unroll
  for (int i = 0; i < 2; ++i) {
    const int flat = tid + i * 256;
    const int drow = flat >> 3;
    const int s8   = (flat & 7) * 8;
    v8bf val;
#pragma unroll
    for (int j = 0; j < 8; ++j) val[j] = tile[s8 + j][drow];
    *(v8bf*)(dst + (size_t)drow * SS + s8) = val;
  }
}

// ---------------------------------------------------------------------------
// Transpose low-rank V: vlr[b][s][d] -> vlrT[b][d][s] (bf16), LDS-tiled.
// ---------------------------------------------------------------------------
__global__ void transpose_vlr_kernel(const __bf16* __restrict__ vlr, __bf16* __restrict__ vlrT) {
  __shared__ __align__(16) __bf16 tile[64][72];
  const int tid = threadIdx.x;
  const int b  = blockIdx.x;
  const int s0 = blockIdx.y * 64;
  const int d0 = blockIdx.z * 64;
  const __bf16* src = vlr + ((size_t)b * SS + s0) * DD + d0;
#pragma unroll
  for (int i = 0; i < 2; ++i) {
    const int flat = tid + i * 256;
    const int srow = flat >> 3;
    const int c8   = (flat & 7) * 8;
    *(v8bf*)(&tile[srow][c8]) = *(const v8bf*)(src + (size_t)srow * DD + c8);
  }
  __syncthreads();
  __bf16* dst = vlrT + ((size_t)b * DD + d0) * SS + s0;
#pragma unroll
  for (int i = 0; i < 2; ++i) {
    const int flat = tid + i * 256;
    const int drow = flat >> 3;
    const int s8   = (flat & 7) * 8;
    v8bf val;
#pragma unroll
    for (int j = 0; j < 8; ++j) val[j] = tile[s8 + j][drow];
    *(v8bf*)(dst + (size_t)drow * SS + s8) = val;
  }
}

// ---------------------------------------------------------------------------
// Dense multi-head attention, flash-style. qkv: [B,S,3D] bf16 (Q,K slices),
// vT: [B*H][dh][S] bf16 (pre-transposed V). One wave per 16 query rows.
// Per 32-key step: 4 WMMAs for scores (dh=64), online softmax (shfl_xor
// half-wave row reductions), P transposed C->A layout via per-wave LDS,
// 4 WMMAs for PV -- all fragment loads contiguous.
// ---------------------------------------------------------------------------
__global__ void dense_attn_kernel(const __bf16* __restrict__ qkv,
                                  const __bf16* __restrict__ vT,
                                  __bf16* __restrict__ ctx) {
  __shared__ __bf16 pbuf[8][16][32];
  const int tid  = threadIdx.x;
  const int w    = tid >> 5;
  const int lane = tid & 31;
  const int l16  = lane & 15;
  const int half = lane >> 4;
  const int chunks = SS / 128;
  const int bh = blockIdx.x / chunks;
  const int qc = blockIdx.x % chunks;
  const int b = bh / HH, h = bh % HH;
  const int q0 = qc * 128 + w * 16;
  const int ld = 3 * DD;
  const __bf16* qp = qkv + (size_t)b * SS * ld + h * DH;
  const __bf16* kp = qp + DD;
  const __bf16* vTh = vT + (size_t)bh * DH * SS;

  const v16bf aq0 = load_a_frag(qp, ld, q0 + l16, 0, half);
  const v16bf aq1 = load_a_frag(qp, ld, q0 + l16, 32, half);

  v8f acc[4] = {};
  float mrow[8], lrow[8];
#pragma unroll
  for (int r = 0; r < 8; ++r) { mrow[r] = -1e30f; lrow[r] = 0.0f; }

  for (int kb = 0; kb < SS; kb += 32) {
    v8f sc0 = {}, sc1 = {};
    {
      v16bf bk;
      bk = load_b_nt(kp, ld, kb + l16, 0, half);
      sc0 = WMMA_BF16(aq0, bk, sc0);
      bk = load_b_nt(kp, ld, kb + l16, 32, half);
      sc0 = WMMA_BF16(aq1, bk, sc0);
      bk = load_b_nt(kp, ld, kb + 16 + l16, 0, half);
      sc1 = WMMA_BF16(aq0, bk, sc1);
      bk = load_b_nt(kp, ld, kb + 16 + l16, 32, half);
      sc1 = WMMA_BF16(aq1, bk, sc1);
    }
    float corr[8];
#pragma unroll
    for (int r = 0; r < 8; ++r) {
      const float v0 = sc0[r] * 0.125f;   // 1/sqrt(64)
      const float v1 = sc1[r] * 0.125f;
      float mx = fmaxf(v0, v1);
#pragma unroll
      for (int m = 8; m >= 1; m >>= 1) mx = fmaxf(mx, __shfl_xor(mx, m, 32));
      const float nm = fmaxf(mrow[r], mx);
      const float p0 = __expf(v0 - nm);
      const float p1 = __expf(v1 - nm);
      float ps = p0 + p1;
#pragma unroll
      for (int m = 8; m >= 1; m >>= 1) ps += __shfl_xor(ps, m, 32);
      const float c = __expf(mrow[r] - nm);
      lrow[r] = lrow[r] * c + ps;
      mrow[r] = nm;
      corr[r] = c;
      pbuf[w][r + half * 8][l16]      = (__bf16)p0;   // C-layout -> LDS
      pbuf[w][r + half * 8][16 + l16] = (__bf16)p1;
    }
#pragma unroll
    for (int t = 0; t < 4; ++t)
#pragma unroll
      for (int r = 0; r < 8; ++r) acc[t][r] *= corr[r];

    v16bf pa;  // A-layout read of P from LDS (same wave; DS ops stay in order)
#pragma unroll
    for (int j = 0; j < 8; ++j) {
      pa[j]     = pbuf[w][l16][half * 8 + j];
      pa[j + 8] = pbuf[w][l16][16 + half * 8 + j];
    }
#pragma unroll
    for (int t = 0; t < 4; ++t) {
      v16bf bv = load_b_nt(vTh, SS, t * 16 + l16, kb, half);  // two 16B loads along S
      acc[t] = WMMA_BF16(pa, bv, acc[t]);
    }
  }

#pragma unroll
  for (int t = 0; t < 4; ++t) {
    const int n = h * DH + t * 16 + l16;
#pragma unroll
    for (int r = 0; r < 8; ++r) {
      const int row = q0 + r + half * 8;
      ctx[((size_t)b * SS + row) * DD + n] = (__bf16)(acc[t][r] / lrow[r]);
    }
  }
}

// ---------------------------------------------------------------------------
// Row softmax + top-k (k=204) sparsify + renormalize. One block per row.
// ---------------------------------------------------------------------------
__global__ void softmax_topk_kernel(const float* __restrict__ sc, __bf16* __restrict__ sp) {
  __shared__ float buf[128];
  const int row = blockIdx.x, tid = threadIdx.x;
  const float* rp = sc + (size_t)row * SS;
  float v[8];
  float mx = -1e30f;
#pragma unroll
  for (int i = 0; i < 8; ++i) { v[i] = rp[tid + i * 128]; mx = fmaxf(mx, v[i]); }
  buf[tid] = mx; __syncthreads();
  for (int s = 64; s > 0; s >>= 1) { if (tid < s) buf[tid] = fmaxf(buf[tid], buf[tid + s]); __syncthreads(); }
  mx = buf[0]; __syncthreads();

  float sum = 0.0f;
#pragma unroll
  for (int i = 0; i < 8; ++i) { v[i] = __expf(v[i] - mx); sum += v[i]; }
  buf[tid] = sum; __syncthreads();
  for (int s = 64; s > 0; s >>= 1) { if (tid < s) buf[tid] += buf[tid + s]; __syncthreads(); }
  const float inv = 1.0f / buf[0]; __syncthreads();
#pragma unroll
  for (int i = 0; i < 8; ++i) v[i] *= inv;

  float lo = 0.0f, hi = 1.0f;
  for (int it = 0; it < 24; ++it) {
    const float mid = 0.5f * (lo + hi);
    float c = 0.0f;
#pragma unroll
    for (int i = 0; i < 8; ++i) c += (v[i] >= mid) ? 1.0f : 0.0f;
    buf[tid] = c; __syncthreads();
    for (int s = 64; s > 0; s >>= 1) { if (tid < s) buf[tid] += buf[tid + s]; __syncthreads(); }
    const float cnt = buf[0]; __syncthreads();
    if (cnt >= (float)KK_TOP) lo = mid; else hi = mid;
  }

  float s2 = 0.0f;
#pragma unroll
  for (int i = 0; i < 8; ++i) if (v[i] >= lo) s2 += v[i];
  buf[tid] = s2; __syncthreads();
  for (int s = 64; s > 0; s >>= 1) { if (tid < s) buf[tid] += buf[tid + s]; __syncthreads(); }
  const float invs = 1.0f / (buf[0] + 1e-9f);
#pragma unroll
  for (int i = 0; i < 8; ++i)
    sp[(size_t)row * SS + tid + i * 128] = (v[i] >= lo) ? (__bf16)(v[i] * invs) : (__bf16)0.0f;
}

// ---------------------------------------------------------------------------
// fused: y = src + sig*dense + (1-sig)*sparse ; x = LN1(y) -> f32 + bf16
// ---------------------------------------------------------------------------
__global__ void fuse_ln1_kernel(const float* __restrict__ src, const float* __restrict__ dense,
                                const float* __restrict__ sparse, const float* __restrict__ lam,
                                const float* __restrict__ g, const float* __restrict__ be,
                                float* __restrict__ xf, __bf16* __restrict__ xb) {
  __shared__ float buf[256];
  const int row = blockIdx.x, tid = threadIdx.x;
  const size_t base = (size_t)row * DD;
  const float sig = 1.0f / (1.0f + __expf(-lam[0]));
  float y[4];
  float s = 0.0f;
#pragma unroll
  for (int i = 0; i < 4; ++i) {
    const int c = tid + i * 256;
    y[i] = src[base + c] + sig * dense[base + c] + (1.0f - sig) * sparse[base + c];
    s += y[i];
  }
  buf[tid] = s; __syncthreads();
  for (int st = 128; st > 0; st >>= 1) { if (tid < st) buf[tid] += buf[tid + st]; __syncthreads(); }
  const float mean = buf[0] * (1.0f / DD); __syncthreads();
  float vv = 0.0f;
#pragma unroll
  for (int i = 0; i < 4; ++i) { const float d = y[i] - mean; vv += d * d; }
  buf[tid] = vv; __syncthreads();
  for (int st = 128; st > 0; st >>= 1) { if (tid < st) buf[tid] += buf[tid + st]; __syncthreads(); }
  const float rstd = rsqrtf(buf[0] * (1.0f / DD) + 1e-5f);
#pragma unroll
  for (int i = 0; i < 4; ++i) {
    const int c = tid + i * 256;
    const float o = (y[i] - mean) * rstd * g[c] + be[c];
    xf[base + c] = o;
    xb[base + c] = (__bf16)o;
  }
}

// ---------------------------------------------------------------------------
// out = LN2(x + ff)
// ---------------------------------------------------------------------------
__global__ void ln2_kernel(const float* __restrict__ x, const float* __restrict__ ff,
                           const float* __restrict__ g, const float* __restrict__ be,
                           float* __restrict__ out) {
  __shared__ float buf[256];
  const int row = blockIdx.x, tid = threadIdx.x;
  const size_t base = (size_t)row * DD;
  float y[4];
  float s = 0.0f;
#pragma unroll
  for (int i = 0; i < 4; ++i) { const int c = tid + i * 256; y[i] = x[base + c] + ff[base + c]; s += y[i]; }
  buf[tid] = s; __syncthreads();
  for (int st = 128; st > 0; st >>= 1) { if (tid < st) buf[tid] += buf[tid + st]; __syncthreads(); }
  const float mean = buf[0] * (1.0f / DD); __syncthreads();
  float vv = 0.0f;
#pragma unroll
  for (int i = 0; i < 4; ++i) { const float d = y[i] - mean; vv += d * d; }
  buf[tid] = vv; __syncthreads();
  for (int st = 128; st > 0; st >>= 1) { if (tid < st) buf[tid] += buf[tid + st]; __syncthreads(); }
  const float rstd = rsqrtf(buf[0] * (1.0f / DD) + 1e-5f);
#pragma unroll
  for (int i = 0; i < 4; ++i) {
    const int c = tid + i * 256;
    out[base + c] = (y[i] - mean) * rstd * g[c] + be[c];
  }
}

// ---------------------------------------------------------------------------
extern "C" void kernel_launch(void* const* d_in, const int* in_sizes, int n_in,
                              void* d_out, int out_size, void* d_ws, size_t ws_size,
                              hipStream_t stream) {
  (void)in_sizes; (void)n_in; (void)out_size; (void)ws_size;
  const float* src   = (const float*)d_in[0];
  const float* w_qkv = (const float*)d_in[1];
  const float* b_qkv = (const float*)d_in[2];
  const float* w_out = (const float*)d_in[3];
  const float* b_out = (const float*)d_in[4];
  const float* w_qp  = (const float*)d_in[5];
  const float* b_qp  = (const float*)d_in[6];
  const float* w_kp  = (const float*)d_in[7];
  const float* b_kp  = (const float*)d_in[8];
  const float* w_vp  = (const float*)d_in[9];
  const float* b_vp  = (const float*)d_in[10];
  const float* lam   = (const float*)d_in[11];
  const float* w_ff1 = (const float*)d_in[12];
  const float* b_ff1 = (const float*)d_in[13];
  const float* w_ff2 = (const float*)d_in[14];
  const float* b_ff2 = (const float*)d_in[15];
  const float* ln1g  = (const float*)d_in[16];
  const float* ln1b  = (const float*)d_in[17];
  const float* ln2g  = (const float*)d_in[18];
  const float* ln2b  = (const float*)d_in[19];
  float* out = (float*)d_out;

  char* ws = (char*)d_ws;
  size_t off = 0;
  auto alloc = [&](size_t bytes) -> char* {
    char* p = ws + off;
    off += (bytes + 255) & ~(size_t)255;
    return p;
  };

  const size_t MS = (size_t)BB * SS;  // 4096 total rows
  __bf16* src_b   = (__bf16*)alloc(MS * DD * 2);
  __bf16* wqkv_b  = (__bf16*)alloc((size_t)3 * DD * DD * 2);
  __bf16* wout_b  = (__bf16*)alloc((size_t)DD * DD * 2);
  __bf16* wqp_b   = (__bf16*)alloc((size_t)RR * DD * 2);
  __bf16* wkp_b   = (__bf16*)alloc((size_t)RR * DD * 2);
  __bf16* wvp_b   = (__bf16*)alloc((size_t)DD * DD * 2);
  __bf16* wff1_b  = (__bf16*)alloc((size_t)DFFX * DD * 2);
  __bf16* wff2_b  = (__bf16*)alloc((size_t)DD * DFFX * 2);
  __bf16* qkv_b   = (__bf16*)alloc(MS * 3 * DD * 2);
  __bf16* vT_b    = (__bf16*)alloc((size_t)BB * HH * DH * SS * 2);
  __bf16* ctx_b   = (__bf16*)alloc(MS * DD * 2);
  float*  dense_f = (float*) alloc(MS * DD * 4);
  __bf16* qlr_b   = (__bf16*)alloc(MS * RR * 2);
  __bf16* klr_b   = (__bf16*)alloc(MS * RR * 2);
  __bf16* vlr_b   = (__bf16*)alloc(MS * DD * 2);
  __bf16* vlrT_b  = (__bf16*)alloc(MS * DD * 2);
  float*  sc2_f   = (float*) alloc((size_t)BB * SS * SS * 4);
  __bf16* sp_b    = (__bf16*)alloc((size_t)BB * SS * SS * 2);
  float*  spo_f   = (float*) alloc(MS * DD * 4);
  float*  x_f     = (float*) alloc(MS * DD * 4);
  __bf16* x_b     = (__bf16*)alloc(MS * DD * 2);
  __bf16* ffh_b   = (__bf16*)alloc(MS * DFFX * 2);
  float*  ff_f    = (float*) alloc(MS * DD * 4);

  auto cvt = [&](const float* in, __bf16* o, size_t n) {
    int blocks = (int)((n + 255) / 256);
    if (blocks > 4096) blocks = 4096;
    cvt_f32_bf16<<<blocks, 256, 0, stream>>>(in, o, (int)n);
  };
  cvt(src,   src_b,  MS * DD);
  cvt(w_qkv, wqkv_b, (size_t)3 * DD * DD);
  cvt(w_out, wout_b, (size_t)DD * DD);
  cvt(w_qp,  wqp_b,  (size_t)RR * DD);
  cvt(w_kp,  wkp_b,  (size_t)RR * DD);
  cvt(w_vp,  wvp_b,  (size_t)DD * DD);
  cvt(w_ff1, wff1_b, (size_t)DFFX * DD);
  cvt(w_ff2, wff2_b, (size_t)DD * DFFX);

  auto gemm = [&](const __bf16* Ap, const __bf16* Wp, const float* bias, float scale,
                  float* Cf, __bf16* Cb, int M, int N, int K,
                  int lda, int ldw, int ldc, int relu) {
    const int waves = (M / 32) * (N / 64);
    const int blocks = (waves + 7) / 8;
    gemm_bf16<<<blocks, 256, 0, stream>>>(Ap, Wp, bias, scale, Cf, Cb,
                                          M, N, K, lda, ldw, ldc, relu);
  };

  // 1) QKV projection -> [B,S,3D] bf16
  gemm(src_b, wqkv_b, b_qkv, 1.0f, nullptr, qkv_b, 4096, 3072, 1024, 1024, 1024, 3072, 0);
  // 2) pre-transpose V per head (async global->LDS staging), then flash attention
  transpose_v_heads<<<dim3(BB * HH, SS / 64), 256, 0, stream>>>(qkv_b, vT_b);
  dense_attn_kernel<<<BB * HH * (SS / 128), 256, 0, stream>>>(qkv_b, vT_b, ctx_b);
  // 3) out projection -> dense_output f32
  gemm(ctx_b, wout_b, b_out, 1.0f, dense_f, nullptr, 4096, 1024, 1024, 1024, 1024, 1024, 0);
  // 4) low-rank projections
  gemm(src_b, wqp_b, b_qp, 1.0f, nullptr, qlr_b, 4096, 64, 1024, 1024, 1024, 64, 0);
  gemm(src_b, wkp_b, b_kp, 1.0f, nullptr, klr_b, 4096, 64, 1024, 1024, 1024, 64, 0);
  gemm(src_b, wvp_b, b_vp, 1.0f, nullptr, vlr_b, 4096, 1024, 1024, 1024, 1024, 1024, 0);
  transpose_vlr_kernel<<<dim3(BB, SS / 64, DD / 64), 256, 0, stream>>>(vlr_b, vlrT_b);
  // 5) low-rank scores: per batch, Q[S,R] x K[S,R]^T / sqrt(R)
  for (int b = 0; b < BB; ++b)
    gemm(qlr_b + (size_t)b * SS * RR, klr_b + (size_t)b * SS * RR, nullptr, 0.125f,
         sc2_f + (size_t)b * SS * SS, nullptr, 1024, 1024, 64, 64, 64, 1024, 0);
  // 6) softmax + top-k sparsify + renormalize -> bf16 sparse attention
  softmax_topk_kernel<<<BB * SS, 128, 0, stream>>>(sc2_f, sp_b);
  // 7) sparse_attn[S,S] x V[S,D] == sparse x (V^T)[D,S]^T : plain NT GEMM now
  for (int b = 0; b < BB; ++b)
    gemm(sp_b + (size_t)b * SS * SS, vlrT_b + (size_t)b * DD * SS, nullptr, 1.0f,
         spo_f + (size_t)b * SS * DD, nullptr, 1024, 1024, 1024, 1024, 1024, 1024, 0);
  // 8) fuse + residual + LN1
  fuse_ln1_kernel<<<BB * SS, 256, 0, stream>>>(src, dense_f, spo_f, lam, ln1g, ln1b, x_f, x_b);
  // 9) FFN
  gemm(x_b, wff1_b, b_ff1, 1.0f, nullptr, ffh_b, 4096, 4096, 1024, 1024, 1024, 4096, 1);
  gemm(ffh_b, wff2_b, b_ff2, 1.0f, ff_f, nullptr, 4096, 1024, 4096, 4096, 4096, 1024, 0);
  // 10) residual + LN2 -> output
  ln2_kernel<<<BB * SS, 256, 0, stream>>>(x_f, ff_f, ln2g, ln2b, out);
}
